// SAModule_18691697672883
// MI455X (gfx1250) — compile-verified, matrix operand order checked by reading, not run on hardware
//
#include <hip/hip_runtime.h>

// ---------------- problem constants ----------------
#define NPTS  16384
#define NIN   128
#define NOUT  256
#define MQ    4096
#define KNB   32
#define DIMP  160           // DIM=131 padded to 5*32 for bf16 WMMA K-steps
#define RAD2  0.005625f     // 0.075^2
#define CANDMAX 256

typedef __attribute__((ext_vector_type(16))) __bf16 bf16x16;
typedef __attribute__((ext_vector_type(8)))  __bf16 bf16x8;
typedef __attribute__((ext_vector_type(8)))  float  f32x8;

// native bf16 convert (v_cvt_*_bf16 on gfx1250; RNE per MODE default)
static __device__ inline __bf16 f2bf(float f) { return (__bf16)f; }

union FragU { bf16x16 v; bf16x8 h[2]; };

// Pack a 16x32 bf16 WMMA A/B fragment from an LDS row-major tile.
// ISA layout (16-bit A 16x32): lane l holds row (l&15); halves 0..7 = K hi*8+0..7,
// halves 8..15 = K 16+hi*8+0..7 with hi = l>>4. B uses the same pattern when the
// weight matrix is stored transposed ([out][in]) so "row" = output column.
static __device__ inline bf16x16 load_frag(const __bf16* base, int stride,
                                           int rowBase, int kBase, int lane) {
  int r  = rowBase + (lane & 15);
  int hi = (lane >> 4) * 8;
  const __bf16* p = base + r * stride + kBase + hi;
  FragU f;
  f.h[0] = *(const bf16x8*)(p);
  f.h[1] = *(const bf16x8*)(p + 16);
  return f.v;
}

// ---------------- 1) farthest point sampling ----------------
__global__ __launch_bounds__(1024) void fps_kernel(const float* __restrict__ pos,
                                                   const int*   __restrict__ batch,
                                                   float* __restrict__ qpos,
                                                   int*   __restrict__ idxOut,
                                                   int*   __restrict__ batchOut) {
  __shared__ float redV[32];
  __shared__ int   redJ[32];
  __shared__ int   bcast;
  const int tid  = threadIdx.x;
  const int lane = tid & 31, wave = tid >> 5;

  // each thread owns 16 contiguous points, fully register-resident
  float px[16], py[16], pz[16], dreg[16];
  const float p0x = pos[0], p0y = pos[1], p0z = pos[2];
#pragma unroll
  for (int e = 0; e < 16; ++e) {
    int j = tid * 16 + e;
    px[e] = pos[3*j]; py[e] = pos[3*j+1]; pz[e] = pos[3*j+2];
    float dx = px[e]-p0x, dy = py[e]-p0y, dz = pz[e]-p0z;
    dreg[e] = dx*dx + dy*dy + dz*dz;
  }
  if (tid == 0) {
    idxOut[0] = 0; batchOut[0] = batch[0];
    qpos[0] = p0x; qpos[1] = p0y; qpos[2] = p0z;
  }

  for (int m = 1; m < MQ; ++m) {
    float bv = -1.0f; int bj = 0;
#pragma unroll
    for (int e = 0; e < 16; ++e)
      if (dreg[e] > bv) { bv = dreg[e]; bj = tid*16 + e; }   // first-max tiebreak
#pragma unroll
    for (int off = 16; off >= 1; off >>= 1) {
      float ov = __shfl_xor(bv, off, 32);
      int   oj = __shfl_xor(bj, off, 32);
      if (ov > bv || (ov == bv && oj < bj)) { bv = ov; bj = oj; }
    }
    if (lane == 0) { redV[wave] = bv; redJ[wave] = bj; }
    __syncthreads();
    if (wave == 0) {
      bv = redV[lane]; bj = redJ[lane];
#pragma unroll
      for (int off = 16; off >= 1; off >>= 1) {
        float ov = __shfl_xor(bv, off, 32);
        int   oj = __shfl_xor(bj, off, 32);
        if (ov > bv || (ov == bv && oj < bj)) { bv = ov; bj = oj; }
      }
      if (lane == 0) {
        bcast = bj;
        idxOut[m] = bj; batchOut[m] = batch[bj];
        qpos[3*m] = pos[3*bj]; qpos[3*m+1] = pos[3*bj+1]; qpos[3*m+2] = pos[3*bj+2];
      }
    }
    __syncthreads();
    const int nxt = bcast;
    const float nx = pos[3*nxt], ny = pos[3*nxt+1], nz = pos[3*nxt+2];
#pragma unroll
    for (int e = 0; e < 16; ++e) {
      float dx = px[e]-nx, dy = py[e]-ny, dz = pz[e]-nz;
      float dn = dx*dx + dy*dy + dz*dz;
      dreg[e] = fminf(dreg[e], dn);
    }
  }
}

// ---------------- 2) radius ball query, K nearest within R ----------------
__global__ __launch_bounds__(256) void ballq_kernel(const float* __restrict__ pos,
                                                    const float* __restrict__ qpos,
                                                    int* __restrict__ nbr,
                                                    int* __restrict__ cnt) {
  __shared__ float cd2[CANDMAX];
  __shared__ int   cj[CANDMAX];
  __shared__ int   used[CANDMAX];
  __shared__ int   counter;
  const int i = blockIdx.x;
  const int tid = threadIdx.x;
  if (tid == 0) counter = 0;
  if (tid < CANDMAX) used[tid] = 0;
  const float qx = qpos[3*i], qy = qpos[3*i+1], qz = qpos[3*i+2];
  __syncthreads();

  for (int j = tid; j < NPTS; j += 256) {
    float dx = pos[3*j]-qx, dy = pos[3*j+1]-qy, dz = pos[3*j+2]-qz;
    float d2 = dx*dx + dy*dy + dz*dz;
    if (d2 <= RAD2) {
      int p = atomicAdd(&counter, 1);
      if (p < CANDMAX) { cd2[p] = d2; cj[p] = j; }
    }
  }
  __syncthreads();
  int c = counter; if (c > CANDMAX) c = CANDMAX;

  if (c <= KNB) {
    // self-edge (d2==0) guarantees c >= 1; pad tail with a valid index (masked later)
    if (tid < KNB) nbr[i*KNB + tid] = (tid < c) ? cj[tid] : cj[0];
    if (tid == 0)  cnt[i] = c;
  } else {
    if (tid < 32) {   // wave 0: K rounds of argmin over the small candidate list
      for (int k = 0; k < KNB; ++k) {
        float bd = 3.0e38f; int bp = -1;
        for (int p = tid; p < c; p += 32)
          if (!used[p] && cd2[p] < bd) { bd = cd2[p]; bp = p; }
#pragma unroll
        for (int off = 16; off >= 1; off >>= 1) {
          float od = __shfl_xor(bd, off, 32);
          int   op = __shfl_xor(bp, off, 32);
          if (od < bd || (od == bd && op >= 0 && (bp < 0 || op < bp))) { bd = od; bp = op; }
        }
        if (tid == 0) { used[bp] = 1; nbr[i*KNB + k] = cj[bp]; }
        __builtin_amdgcn_wave_barrier();   // keep used[] store/load order
      }
      if (tid == 0) cnt[i] = KNB;
    }
  }
}

// ---------------- 3) fused PointConv MLP (bf16 WMMA) + max-aggregate ----------------
// Dynamic LDS layout (bytes):
//   [0,51200)        W1^T  bf16 [160][160]
//   [51200,133120)   W2^T  bf16 [256][160]
//   [133120,133760)  b1 f32 [160]
//   [133760,134784)  b2 f32 [256]
//   [134784, +wid*20480) per-wave: msg bf16 [32][160], h1 bf16 [32][160]
#define SMEM3_BYTES (134784 + 4 * 20480)

__global__ __launch_bounds__(128) void pointconv_kernel(
    const float* __restrict__ x,   const float* __restrict__ pos,
    const float* __restrict__ qpos,
    const float* __restrict__ W1,  const float* __restrict__ b1,
    const float* __restrict__ W2,  const float* __restrict__ b2,
    const int*   __restrict__ nbr, const int*   __restrict__ cnt,
    float* __restrict__ out) {
  extern __shared__ char smem[];
  __bf16* w1t = (__bf16*)smem;
  __bf16* w2t = (__bf16*)(smem + 51200);
  float*  b1s = (float*)(smem + 133120);
  float*  b2s = (float*)(smem + 133760);
  const int tid  = threadIdx.x;
  const int lane = tid & 31, wid = tid >> 5;
  __bf16* msg = (__bf16*)(smem + 134784 + wid * 20480);
  __bf16* h1  = msg + 32 * DIMP;

  // transpose + pad weights into LDS as bf16 (131 -> 160 zero-padded)
  for (int t = tid; t < 160*160; t += 128) {
    int o = t / DIMP, ii = t % DIMP;
    float v = (o < 131 && ii < 131) ? W1[ii*131 + o] : 0.0f;
    w1t[t] = f2bf(v);
  }
  for (int t = tid; t < 256*160; t += 128) {
    int o = t / DIMP, ii = t % DIMP;
    float v = (ii < 131) ? W2[ii*256 + o] : 0.0f;
    w2t[t] = f2bf(v);
  }
  for (int t = tid; t < 160; t += 128) b1s[t] = (t < 131) ? b1[t] : 0.0f;
  for (int t = tid; t < 256; t += 128) b2s[t] = b2[t];
  __syncthreads();

  const int ci = blockIdx.x * 4 + wid;       // one centroid per wave

  // gather msg row (lane = neighbor slot): [x_j (128) | rel (3) | zeros -> 160]
  const int j = nbr[ci*KNB + lane];
  const float qx = qpos[3*ci], qy = qpos[3*ci+1], qz = qpos[3*ci+2];
  const float4* xr = (const float4*)(x + (long)j * NIN);
#pragma unroll
  for (int cbase = 0; cbase < NIN; cbase += 8) {
    float4 a = xr[cbase/4], b = xr[cbase/4 + 1];
    bf16x8 h;
    h[0] = f2bf(a.x); h[1] = f2bf(a.y); h[2] = f2bf(a.z); h[3] = f2bf(a.w);
    h[4] = f2bf(b.x); h[5] = f2bf(b.y); h[6] = f2bf(b.z); h[7] = f2bf(b.w);
    *(bf16x8*)(msg + lane*DIMP + cbase) = h;
  }
  {
    bf16x8 t = {};
    t[0] = f2bf(pos[3*j]   - qx);
    t[1] = f2bf(pos[3*j+1] - qy);
    t[2] = f2bf(pos[3*j+2] - qz);
    *(bf16x8*)(msg + lane*DIMP + 128) = t;
    bf16x8 z = {};
    *(bf16x8*)(msg + lane*DIMP + 136) = z;
    *(bf16x8*)(msg + lane*DIMP + 144) = z;
    *(bf16x8*)(msg + lane*DIMP + 152) = z;
  }
  __builtin_amdgcn_wave_barrier();   // msg/h1 are wave-private; DS is in-order per wave

  int c = cnt[ci]; if (c > KNB) c = KNB;
  const int col = lane & 15, hi8 = (lane >> 4) * 8;

  // layer 1: h1 = relu(msg @ W1 + b1)   (32x160 = 2x10 tiles, 5 K-steps each)
  // A fragments depend only on (mt,kc): keep the 5 of them in registers per M-tile.
  for (int mt = 0; mt < 2; ++mt) {
    bf16x16 afr[5];
#pragma unroll
    for (int kk = 0; kk < 5; ++kk)
      afr[kk] = load_frag(msg, DIMP, mt*16, kk*32, lane);
    for (int nt = 0; nt < 10; ++nt) {
      f32x8 acc = {};
#pragma unroll
      for (int kk = 0; kk < 5; ++kk) {
        bf16x16 b = load_frag(w1t, DIMP, nt*16, kk*32, lane);
        acc = __builtin_amdgcn_wmma_f32_16x16x32_bf16(false, afr[kk], false, b,
                                                      (short)0, acc, false, false);
      }
      float bias = b1s[nt*16 + col];
#pragma unroll
      for (int r = 0; r < 8; ++r) {
        float v = fmaxf(acc[r] + bias, 0.0f);
        h1[(mt*16 + r + hi8)*DIMP + nt*16 + col] = f2bf(v);
      }
    }
  }
  __builtin_amdgcn_wave_barrier();

  // layer 2 + masked max over neighbors: out[ci] = max_k relu(h1 @ W2 + b2)
  // mt outer so the 5 A fragments (from h1) load once per M-tile; colmax[16] in regs.
  float colmax[16];
#pragma unroll
  for (int nt = 0; nt < 16; ++nt) colmax[nt] = -3.0e38f;

  for (int mt = 0; mt < 2; ++mt) {
    bf16x16 afr[5];
#pragma unroll
    for (int kk = 0; kk < 5; ++kk)
      afr[kk] = load_frag(h1, DIMP, mt*16, kk*32, lane);
#pragma unroll
    for (int nt = 0; nt < 16; ++nt) {
      f32x8 acc = {};
#pragma unroll
      for (int kk = 0; kk < 5; ++kk) {
        bf16x16 b = load_frag(w2t, DIMP, nt*16, kk*32, lane);
        acc = __builtin_amdgcn_wmma_f32_16x16x32_bf16(false, afr[kk], false, b,
                                                      (short)0, acc, false, false);
      }
      float bias = b2s[nt*16 + col];
#pragma unroll
      for (int r = 0; r < 8; ++r) {
        int row = mt*16 + r + hi8;         // neighbor index of this D element
        float v = fmaxf(acc[r] + bias, 0.0f);
        if (row >= c) v = -3.0e38f;        // invalid neighbor
        colmax[nt] = fmaxf(colmax[nt], v);
      }
    }
  }
#pragma unroll
  for (int nt = 0; nt < 16; ++nt) {
    float o = __shfl_xor(colmax[nt], 16, 32);  // fold rows 8..15 / 24..31 halves
    float v = fmaxf(colmax[nt], o);
    if (lane < 16) out[(long)ci*NOUT + nt*16 + lane] = v;
  }
}

// ---------------- host launcher ----------------
extern "C" void kernel_launch(void* const* d_in, const int* in_sizes, int n_in,
                              void* d_out, int out_size, void* d_ws, size_t ws_size,
                              hipStream_t stream) {
  (void)in_sizes; (void)n_in; (void)out_size; (void)ws_size;
  const float* x     = (const float*)d_in[0];
  const float* pos   = (const float*)d_in[1];
  const int*   batch = (const int*)  d_in[2];
  const float* W1    = (const float*)d_in[3];
  const float* b1    = (const float*)d_in[4];
  const float* W2    = (const float*)d_in[5];
  const float* b2    = (const float*)d_in[6];

  // d_out layout (reference tuple, flat): out[M,256] | qpos[M,3] | batch[M] | idx[M]
  float* out      = (float*)d_out;
  float* qpos     = out + (size_t)MQ * NOUT;
  int*   batchOut = (int*)(qpos + (size_t)MQ * 3);
  int*   idxOut   = batchOut + MQ;

  // workspace: nbr[M,K] int | cnt[M] int  (~540 KB)
  int* nbr = (int*)d_ws;
  int* cnt = nbr + (size_t)MQ * KNB;

  fps_kernel  <<<1,      1024, 0, stream>>>(pos, batch, qpos, idxOut, batchOut);
  ballq_kernel<<<MQ,      256, 0, stream>>>(pos, qpos, nbr, cnt);
  pointconv_kernel<<<MQ / 4, 128, SMEM3_BYTES, stream>>>(x, pos, qpos,
                                                         W1, b1, W2, b2,
                                                         nbr, cnt, out);
}
